// TransitionUp_24120536334934
// MI455X (gfx1250) — matrix operand order repaired and checked by reading, not compiled, not run on hardware
//
#include <hip/hip_runtime.h>

typedef __attribute__((ext_vector_type(16))) __bf16 v16bf;
typedef __attribute__((ext_vector_type(8)))  float  v8f;

static constexpr int NPTS = 16384;
static constexpr int NSUB = 4096;
static constexpr int CIN  = 512;
static constexpr int COUT = 256;
static constexpr float KNN_EPS = 1e-16f;

// ---------------------------------------------------------------------------
// Repack a row-major K x N fp32 weight matrix into WMMA B-fragment order,
// split into bf16 hi + residual-lo planes.
// Fragment layout chosen here (B operand, wave32; contiguous k, A fragment uses the same):
//   lane L: n = nt*16 + (L&15), half = L>>4; element e (0..15): k = kt*32 + 16*half + e
// Packed address: ((kt*NT + nt)*32 + lane)*16 + e  == flat tid.
// ---------------------------------------------------------------------------
__global__ void pack_b_kernel(const float* __restrict__ W,
                              __bf16* __restrict__ hi,
                              __bf16* __restrict__ lo,
                              int K, int N) {
  int tid = blockIdx.x * blockDim.x + threadIdx.x;
  if (tid >= K * N) return;
  int e    = tid & 15;
  int lane = (tid >> 4) & 31;
  int frag = tid >> 9;
  int NT = N >> 4;
  int nt = frag % NT;
  int kt = frag / NT;
  int k = kt * 32 + ((lane >> 4) << 4) + e;
  int n = nt * 16 + (lane & 15);
  float v = W[(size_t)k * N + n];
  __bf16 h = (__bf16)v;
  hi[tid] = h;
  lo[tid] = (__bf16)(v - (float)h);
}

// ---------------------------------------------------------------------------
// C = relu(A @ B + bias). A: MxK fp32 row-major. B: packed bf16 hi/lo planes.
// One wave -> 16(M) x 64(N) tile: 4 accumulators share one A fragment.
// bf16x3 split per K-step of 32: ahi*bhi + alo*bhi + ahi*blo (fp32 accum).
// ---------------------------------------------------------------------------
__global__ void gemm_bias_relu(const float* __restrict__ A,
                               const __bf16* __restrict__ Bhi,
                               const __bf16* __restrict__ Blo,
                               const float* __restrict__ bias,
                               float* __restrict__ C,
                               int M, int N, int K) {
  const int lane = threadIdx.x & 31;
  const int wid  = (blockIdx.x * blockDim.x + threadIdx.x) >> 5;
  const int nGroups = N >> 6;               // 64-wide N groups
  const int tm   = wid / nGroups;           // 16-row tile index
  const int ng   = wid % nGroups;
  const int half = lane >> 4;
  const int mrow = tm * 16 + (lane & 15);   // A row this lane feeds
  const int NT   = N >> 4;
  const int KT   = K >> 5;

  v8f acc[4] = {v8f{}, v8f{}, v8f{}, v8f{}};
  const float* arow = A + (size_t)mrow * K;

  for (int kt = 0; kt < KT; ++kt) {
    // A fragment: contiguous order (matches packed B) -> lane needs K = kt*32 + {16h..16h+15}
    const float* ab = arow + kt * 32 + half * 16;
    float4 a0 = *(const float4*)(ab);
    float4 a1 = *(const float4*)(ab + 4);
    float4 a2 = *(const float4*)(ab + 8);
    float4 a3 = *(const float4*)(ab + 12);
    float af[16] = {a0.x, a0.y, a0.z, a0.w, a1.x, a1.y, a1.z, a1.w,
                    a2.x, a2.y, a2.z, a2.w, a3.x, a3.y, a3.z, a3.w};
    v16bf ahi, alo;
#pragma unroll
    for (int e = 0; e < 16; ++e) {
      __bf16 h = (__bf16)af[e];
      ahi[e] = h;
      alo[e] = (__bf16)(af[e] - (float)h);
    }
#pragma unroll
    for (int j = 0; j < 4; ++j) {
      const int nt = ng * 4 + j;
      const size_t fbase = ((size_t)(kt * NT + nt) * 32 + lane) * 16;
      v16bf bhi = *(const v16bf*)(Bhi + fbase);   // 32B aligned -> 2x b128
      v16bf blo = *(const v16bf*)(Blo + fbase);
      acc[j] = __builtin_amdgcn_wmma_f32_16x16x32_bf16(
          false, ahi, false, bhi, (short)0, acc[j], false, false);
      acc[j] = __builtin_amdgcn_wmma_f32_16x16x32_bf16(
          false, alo, false, bhi, (short)0, acc[j], false, false);
      acc[j] = __builtin_amdgcn_wmma_f32_16x16x32_bf16(
          false, ahi, false, blo, (short)0, acc[j], false, false);
    }
  }

  // Epilogue: C/D layout -> VGPR r holds row (tm*16 + 8*half + r), col = lane&15.
#pragma unroll
  for (int j = 0; j < 4; ++j) {
    const int col = (ng * 4 + j) * 16 + (lane & 15);
    const float bcol = bias[col];
#pragma unroll
    for (int r = 0; r < 8; ++r) {
      const int row = tm * 16 + half * 8 + r;
      float v = acc[j][r] + bcol;
      C[(size_t)row * N + col] = v > 0.f ? v : 0.f;
    }
  }
}

// ---------------------------------------------------------------------------
// kNN(k=3) with inverse-squared-distance weights.
// Block = 256 threads = 8 waves; pos_sub (48KB) staged in LDS once per block;
// each wave owns one query point, per-lane top-3 then shfl_xor butterfly merge.
// ---------------------------------------------------------------------------
static __device__ __forceinline__ void top3_insert(float d, int i,
                                                   float& d0, int& i0,
                                                   float& d1, int& i1,
                                                   float& d2, int& i2) {
  if (d < d2) {
    if (d < d1) {
      d2 = d1; i2 = i1;
      if (d < d0) { d1 = d0; i1 = i0; d0 = d; i0 = i; }
      else        { d1 = d;  i1 = i; }
    } else      { d2 = d;  i2 = i; }
  }
}

__global__ void knn3_kernel(const float* __restrict__ pos,
                            const float* __restrict__ pos_sub,
                            int* __restrict__ idx_out,
                            float* __restrict__ w_out) {
  __shared__ float sp[NSUB * 3];
  {
    float4* dst = (float4*)sp;
    const float4* src = (const float4*)pos_sub;
    for (int i = threadIdx.x; i < (NSUB * 3) / 4; i += blockDim.x) dst[i] = src[i];
  }
  __syncthreads();

  const int lane = threadIdx.x & 31;
  const int wave = threadIdx.x >> 5;
  const int m = blockIdx.x * 8 + wave;

  const float px = pos[m * 3 + 0];
  const float py = pos[m * 3 + 1];
  const float pz = pos[m * 3 + 2];

  float d0 = 1e30f, d1 = 1e30f, d2 = 1e30f;
  int   i0 = 0,     i1 = 0,     i2 = 0;

  for (int j = lane; j < NSUB; j += 32) {
    float dx = sp[j * 3 + 0] - px;
    float dy = sp[j * 3 + 1] - py;
    float dz = sp[j * 3 + 2] - pz;
    float d = dx * dx + dy * dy + dz * dz;
    top3_insert(d, j, d0, i0, d1, i1, d2, i2);
  }

  // Butterfly all-reduce of sorted triples across the wave32.
  for (int off = 1; off < 32; off <<= 1) {
    float e0 = __shfl_xor(d0, off, 32);
    float e1 = __shfl_xor(d1, off, 32);
    float e2 = __shfl_xor(d2, off, 32);
    int   j0 = __shfl_xor(i0, off, 32);
    int   j1 = __shfl_xor(i1, off, 32);
    int   j2 = __shfl_xor(i2, off, 32);
    top3_insert(e0, j0, d0, i0, d1, i1, d2, i2);
    top3_insert(e1, j1, d0, i0, d1, i1, d2, i2);
    top3_insert(e2, j2, d0, i0, d1, i1, d2, i2);
  }

  if (lane == 0) {
    float w0 = 1.f / fmaxf(d0, KNN_EPS);
    float w1 = 1.f / fmaxf(d1, KNN_EPS);
    float w2 = 1.f / fmaxf(d2, KNN_EPS);
    float inv = 1.f / (w0 + w1 + w2);
    idx_out[m * 3 + 0] = i0;
    idx_out[m * 3 + 1] = i1;
    idx_out[m * 3 + 2] = i2;
    w_out[m * 3 + 0] = w0 * inv;
    w_out[m * 3 + 1] = w1 * inv;
    w_out[m * 3 + 2] = w2 * inv;
  }
}

// ---------------------------------------------------------------------------
// out[m, :] += sum_k w[m,k] * h_sub[idx[m,k], :]   (coalesced row gathers)
// ---------------------------------------------------------------------------
__global__ void interp_add_kernel(const float* __restrict__ h_sub,
                                  const int* __restrict__ idx,
                                  const float* __restrict__ w,
                                  float* __restrict__ out) {
  const int m = blockIdx.x;
  const int c = threadIdx.x;
  const int i0 = idx[m * 3 + 0], i1 = idx[m * 3 + 1], i2 = idx[m * 3 + 2];
  const float w0 = w[m * 3 + 0], w1 = w[m * 3 + 1], w2 = w[m * 3 + 2];
  const size_t o = (size_t)m * COUT + c;
  out[o] = out[o]
         + w0 * h_sub[(size_t)i0 * COUT + c]
         + w1 * h_sub[(size_t)i1 * COUT + c]
         + w2 * h_sub[(size_t)i2 * COUT + c];
}

// ---------------------------------------------------------------------------
extern "C" void kernel_launch(void* const* d_in, const int* in_sizes, int n_in,
                              void* d_out, int out_size, void* d_ws, size_t ws_size,
                              hipStream_t stream) {
  const float* x       = (const float*)d_in[0];  // [16384, 256]
  const float* x_sub   = (const float*)d_in[1];  // [4096, 512]
  const float* pos     = (const float*)d_in[2];  // [16384, 3]
  const float* pos_sub = (const float*)d_in[3];  // [4096, 3]
  const float* W_sub   = (const float*)d_in[4];  // [512, 256]
  const float* b_sub   = (const float*)d_in[5];  // [256]
  const float* W       = (const float*)d_in[6];  // [256, 256]
  const float* b       = (const float*)d_in[7];  // [256]
  float* out = (float*)d_out;                    // [16384, 256]

  char* ws = (char*)d_ws;
  float*  h_sub  = (float*)ws;   ws += (size_t)NSUB * COUT * sizeof(float);
  __bf16* WsubHi = (__bf16*)ws;  ws += (size_t)CIN * COUT * sizeof(__bf16);
  __bf16* WsubLo = (__bf16*)ws;  ws += (size_t)CIN * COUT * sizeof(__bf16);
  __bf16* WHi    = (__bf16*)ws;  ws += (size_t)COUT * COUT * sizeof(__bf16);
  __bf16* WLo    = (__bf16*)ws;  ws += (size_t)COUT * COUT * sizeof(__bf16);
  int*    knnIdx = (int*)ws;     ws += (size_t)NPTS * 3 * sizeof(int);
  float*  knnW   = (float*)ws;

  // 1) pack weights into WMMA B-fragment bf16 hi/lo planes
  pack_b_kernel<<<(CIN * COUT) / 256, 256, 0, stream>>>(W_sub, WsubHi, WsubLo, CIN, COUT);
  pack_b_kernel<<<(COUT * COUT) / 256, 256, 0, stream>>>(W, WHi, WLo, COUT, COUT);

  // 2) h_sub = relu(x_sub @ W_sub + b_sub)
  {
    const int waves = (NSUB / 16) * (COUT / 64);
    gemm_bias_relu<<<waves * 32 / 256, 256, 0, stream>>>(
        x_sub, WsubHi, WsubLo, b_sub, h_sub, NSUB, COUT, CIN);
  }

  // 3) top-3 neighbors + inverse-d2 weights
  knn3_kernel<<<NPTS / 8, 256, 0, stream>>>(pos, pos_sub, knnIdx, knnW);

  // 4) out = relu(x @ W + b)
  {
    const int waves = (NPTS / 16) * (COUT / 64);
    gemm_bias_relu<<<waves * 32 / 256, 256, 0, stream>>>(
        x, WHi, WLo, b, out, NPTS, COUT, COUT);
  }

  // 5) out += knn-interpolated h_sub
  interp_add_kernel<<<NPTS, 256, 0, stream>>>(h_sub, knnIdx, knnW, out);
}
